// Decoder_7387343749145
// MI455X (gfx1250) — matrix operand order repaired and checked by reading, not compiled, not run on hardware
//
#include <hip/hip_runtime.h>
#include <hip/hip_bf16.h>
#include <math.h>

#define B 128
#define S 128
#define H 1024
#define E 512
#define V 32000

typedef __attribute__((ext_vector_type(16))) __bf16 v16bf;
typedef __attribute__((ext_vector_type(8)))  __bf16 v8bf;
typedef __attribute__((ext_vector_type(8)))  float  v8f;
typedef __attribute__((ext_vector_type(4)))  float  v4f;

// ---- fragment helpers (bf16 16x16x32 WMMA, wave32) ---------------------
static __device__ inline v16bf pack16(v4f x0, v4f x1, v4f x2, v4f x3) {
  v16bf f;
#pragma unroll
  for (int i = 0; i < 4; ++i) {
    f[i]      = (__bf16)x0[i];
    f[i + 4]  = (__bf16)x1[i];
    f[i + 8]  = (__bf16)x2[i];
    f[i + 12] = (__bf16)x3[i];
  }
  return f;
}

// A-matrix 16x32 (MxK) from row-major f32: lanes 0-15 -> row M=lane,
// K={0..7,16..23}; lanes 16-31 -> row M=lane-16, K={8..15,24..31}.
static __device__ inline v16bf load_a16x32_f32(const float* __restrict__ base, int ld, int lane) {
  const int r = lane & 15, hi = (lane >> 4) & 1;
  const float* p = base + (size_t)r * ld + hi * 8;
  return pack16(*(const v4f*)p, *(const v4f*)(p + 4),
                *(const v4f*)(p + 16), *(const v4f*)(p + 20));
}

// Same A layout, bf16 source (two 16B loads per lane).
static __device__ inline v16bf load_a16x32_bf(const __bf16* __restrict__ base, int ld, int lane) {
  const int r = lane & 15, hi = (lane >> 4) & 1;
  const __bf16* p = base + (size_t)r * ld + hi * 8;
  const v8bf y0 = *(const v8bf*)p;
  const v8bf y1 = *(const v8bf*)(p + 16);
  v16bf f;
#pragma unroll
  for (int i = 0; i < 8; ++i) { f[i] = y0[i]; f[i + 8] = y1[i]; }
  return f;
}

// B-matrix 32x16 (KxN) from a row-major [N,K] f32 weight slab (B = W^T):
// lanes 0-15 -> column N=lane, K={0..7,16..23}; lanes 16-31 -> K={8..15,24..31}.
static __device__ inline v16bf load_bt32x16_f32(const float* __restrict__ base, int ld, int lane) {
  const int n = lane & 15, kb = ((lane >> 4) & 1) << 3;
  const float* p = base + (size_t)n * ld + kb;
  return pack16(*(const v4f*)p, *(const v4f*)(p + 4),
                *(const v4f*)(p + 16), *(const v4f*)(p + 20));
}

static __device__ inline v8f wmma_bf16(v16bf a, v16bf b, v8f c) {
  return __builtin_amdgcn_wmma_f32_16x16x32_bf16(false, a, false, b, (short)0, c, false, false);
}

// C/D f32 16x16: lane -> N=lane&15 (lane half selects M 0-7 / 8-15), VGPR i -> M=8*(lane>>4)+i.

// ---- Kernel 0: hidden f32 -> bf16 copy ---------------------------------
__global__ void k_cvt_hidden(const float* __restrict__ hidden, __bf16* __restrict__ hidden_bf) {
  const int idx = blockIdx.x * blockDim.x + threadIdx.x;
  if (idx < B * H) hidden_bf[idx] = (__bf16)hidden[idx];
}

// ---- Kernel 1: proj[b,n] = h_last[b,:] . attn_W[n,:H] + attn_b[n] ------
__global__ void k_proj(const float* __restrict__ hidden,
                       const float* __restrict__ attn_W,
                       const float* __restrict__ attn_b,
                       float* __restrict__ proj) {
  const int lane = threadIdx.x & 31;
  const int wave = threadIdx.x >> 5;
  const int mt = blockIdx.x;              // 0..7
  const int nt = blockIdx.y * 8 + wave;   // 0..63
  v8f acc = {};
  const float* ab = hidden + (size_t)mt * 16 * H;
  const float* bb = attn_W + (size_t)nt * 16 * (2 * H);     // W1 = attn_W[:, :H]
  for (int k = 0; k < H; k += 32)
    acc = wmma_bf16(load_a16x32_f32(ab + k, H, lane),
                    load_bt32x16_f32(bb + k, 2 * H, lane), acc);
  const int nl = lane & 15;
  const int mh = (lane >> 4) * 8;
  const int n = nt * 16 + nl;
#pragma unroll
  for (int i = 0; i < 8; ++i)
    proj[(size_t)(mt * 16 + mh + i) * H + n] = acc[i] + attn_b[n];
}

// ---- Kernel 2: fused energy GEMM + tanh + v-dot -> raw scores ----------
__global__ void k_score(const float* __restrict__ enc,
                        const float* __restrict__ attn_W,
                        const float* __restrict__ proj,
                        const float* __restrict__ vvec,
                        float* __restrict__ scores) {
  __shared__ float sh[16];
  const int lane = threadIdx.x & 31;
  const int wave = threadIdx.x >> 5;
  const int b  = blockIdx.x;
  const int st = blockIdx.y;              // 0..7
  if (threadIdx.x < 16) sh[threadIdx.x] = 0.f;
  __syncthreads();
  const float* abase = enc + (size_t)(b * S + st * 16) * H;
  v8f acc[8];
#pragma unroll
  for (int t = 0; t < 8; ++t) acc[t] = (v8f){};
  for (int k = 0; k < H; k += 32) {
    const v16bf a = load_a16x32_f32(abase + k, H, lane);
#pragma unroll
    for (int t = 0; t < 8; ++t) {
      const int nt = wave + t * 8;        // N-tile 0..63
      const float* bbase = attn_W + (size_t)(nt * 16) * (2 * H) + H;  // W2
      acc[t] = wmma_bf16(a, load_bt32x16_f32(bbase + k, 2 * H, lane), acc[t]);
    }
  }
  const int nl = lane & 15;
  const int mh = (lane >> 4) * 8;
  float part[8];
#pragma unroll
  for (int i = 0; i < 8; ++i) part[i] = 0.f;
#pragma unroll
  for (int t = 0; t < 8; ++t) {
    const int n = (wave + t * 8) * 16 + nl;
    const float pv = proj[(size_t)b * H + n];   // includes attn_b
    const float vn = vvec[n];
#pragma unroll
    for (int i = 0; i < 8; ++i) part[i] += vn * tanhf(acc[t][i] + pv);
  }
#pragma unroll
  for (int i = 0; i < 8; ++i) atomicAdd(&sh[mh + i], part[i]);     // ds_add_f32
  __syncthreads();
  if (threadIdx.x < 16) scores[(size_t)b * S + st * 16 + threadIdx.x] = sh[threadIdx.x];
}

// ---- Kernel 3: softmax over S, context, emb gather; emits bf16 acts ----
__global__ void k_softmax_ctx(const float* __restrict__ scores,
                              const float* __restrict__ enc,
                              const int*   __restrict__ x,
                              const float* __restrict__ emb_table,
                              float* __restrict__ attn_out,
                              __bf16* __restrict__ xin_bf,
                              __bf16* __restrict__ outcat_bf) {
  __shared__ float w[S];
  __shared__ float red[64];
  const int b = blockIdx.x;
  const int t = threadIdx.x;              // 0..255
  if (t < S) w[t] = scores[(size_t)b * S + t];
  __syncthreads();
  if (t < 64) red[t] = fmaxf(w[t], w[t + 64]);
  __syncthreads();
  for (int s2 = 32; s2 > 0; s2 >>= 1) {
    if (t < s2) red[t] = fmaxf(red[t], red[t + s2]);
    __syncthreads();
  }
  const float mx = red[0];
  __syncthreads();
  if (t < S) w[t] = expf(w[t] - mx);
  __syncthreads();
  if (t < 64) red[t] = w[t] + w[t + 64];
  __syncthreads();
  for (int s2 = 32; s2 > 0; s2 >>= 1) {
    if (t < s2) red[t] += red[t + s2];
    __syncthreads();
  }
  const float inv = 1.f / red[0];
  __syncthreads();
  if (t < S) {
    w[t] *= inv;
    attn_out[(size_t)b * S + t] = w[t];
  }
  __syncthreads();
#pragma unroll
  for (int j = 0; j < 4; ++j) {
    const int h = t + j * 256;
    float c = 0.f;
    for (int s = 0; s < S; ++s) c += w[s] * enc[((size_t)b * S + s) * H + h];
    xin_bf[(size_t)b * (E + H) + E + h]  = (__bf16)c;
    outcat_bf[(size_t)b * (2 * H) + H + h] = (__bf16)c;
  }
  const int row = x[b];
#pragma unroll
  for (int j = 0; j < 2; ++j) {
    const int e = t + j * 256;
    xin_bf[(size_t)b * (E + H) + e] = (__bf16)emb_table[(size_t)row * E + e];
  }
}

// ---- Kernel 4: gates GEMM; weights streamed once (acc over all 8 M-tiles)
__global__ void k_gates(const __bf16* __restrict__ xin_bf,
                        const __bf16* __restrict__ hidden_bf,
                        const float* __restrict__ W_ih,
                        const float* __restrict__ W_hh,
                        const float* __restrict__ b_ih,
                        const float* __restrict__ b_hh,
                        float* __restrict__ gates) {
  const int lane = threadIdx.x & 31;
  const int wave = threadIdx.x >> 5;
  const int nt = blockIdx.x * 8 + wave;   // 0..255 (N = 4096)
  const int KX = E + H;                   // 1536
  v8f acc[8];
#pragma unroll
  for (int m = 0; m < 8; ++m) acc[m] = (v8f){};
  const float* bx = W_ih + (size_t)nt * 16 * KX;
  for (int k = 0; k < KX; k += 32) {
    const v16bf bfr = load_bt32x16_f32(bx + k, KX, lane);
#pragma unroll
    for (int m = 0; m < 8; ++m)
      acc[m] = wmma_bf16(load_a16x32_bf(xin_bf + (size_t)m * 16 * KX + k, KX, lane), bfr, acc[m]);
  }
  const float* bh = W_hh + (size_t)nt * 16 * H;
  for (int k = 0; k < H; k += 32) {
    const v16bf bfr = load_bt32x16_f32(bh + k, H, lane);
#pragma unroll
    for (int m = 0; m < 8; ++m)
      acc[m] = wmma_bf16(load_a16x32_bf(hidden_bf + (size_t)m * 16 * H + k, H, lane), bfr, acc[m]);
  }
  const int nl = lane & 15;
  const int mh = (lane >> 4) * 8;
  const int n = nt * 16 + nl;
  const float bias = b_ih[n] + b_hh[n];
#pragma unroll
  for (int m = 0; m < 8; ++m)
#pragma unroll
    for (int i = 0; i < 8; ++i)
      gates[(size_t)(m * 16 + mh + i) * (4 * H) + n] = acc[m][i] + bias;
}

// ---- Kernel 5: elementwise LSTM cell -----------------------------------
__global__ void k_lstm(const float* __restrict__ gates,
                       const float* __restrict__ cell,
                       float* __restrict__ h_out,
                       float* __restrict__ c_out,
                       __bf16* __restrict__ outcat_bf) {
  const int idx = blockIdx.x * blockDim.x + threadIdx.x;
  if (idx >= B * H) return;
  const int b = idx / H, h = idx % H;
  const float* g = gates + (size_t)b * 4 * H;
  const float ig = g[h], fg = g[H + h], gg = g[2 * H + h], og = g[3 * H + h];
  const float si = 1.f / (1.f + expf(-ig));
  const float sf = 1.f / (1.f + expf(-fg));
  const float so = 1.f / (1.f + expf(-og));
  const float cn = sf * cell[idx] + si * tanhf(gg);
  const float hn = so * tanhf(cn);
  c_out[idx] = cn;
  h_out[idx] = hn;
  outcat_bf[(size_t)b * 2 * H + h] = (__bf16)hn;
}

// ---- Kernel 6: logits GEMM; fc_W async-staged through LDS --------------
// Each wave owns one N-tile (16 rows of fc_W) and 8 M-tile accumulators.
// The 16x32 f32 B-slab for the NEXT k-step is DMA'd into a per-wave LDS
// double buffer with GLOBAL_LOAD_ASYNC_TO_LDS_B128 (ASYNCcnt) while the
// current slab feeds 8 WMMAs. Async loads complete in order, so
// s_wait_asynccnt 4 (with 8 outstanding) guarantees the current phase.
__global__ void k_logits(const __bf16* __restrict__ outcat_bf,
                         const float* __restrict__ fc_W,
                         const float* __restrict__ fc_b,
                         float* __restrict__ logits) {
  __shared__ alignas(16) float stage[8][2][16 * 32];   // 32 KB
  const int lane = threadIdx.x & 31;
  const int wave = threadIdx.x >> 5;
  const int nt = blockIdx.x * 8 + wave;   // 0..1999 (N = 32000)
  const int K = 2 * H;

  const float* bb = fc_W + (size_t)nt * 16 * K;        // 16 x K tile, row-major
  // chunk ci = j*32+lane covers tile bytes [ci*16, ci*16+16):
  //   row = ci>>3 (global row stride K*4B), cpos = ci&7 (16B within row slab)
  const unsigned lds0 = (unsigned)(uintptr_t)&stage[wave][0][0];
  unsigned goff[4], loff[4];
#pragma unroll
  for (int j = 0; j < 4; ++j) {
    const int ci = j * 32 + lane;
    goff[j] = (unsigned)((ci >> 3) * (K * 4) + (ci & 7) * 16);
    loff[j] = lds0 + (unsigned)(ci * 16);
  }

  v8f acc[8];
#pragma unroll
  for (int m = 0; m < 8; ++m) acc[m] = (v8f){};

  // prologue: stage k=0 into phase 0
#pragma unroll
  for (int j = 0; j < 4; ++j)
    asm volatile("global_load_async_to_lds_b128 %0, %1, %2"
                 :: "v"(loff[j]), "v"(goff[j]), "s"(bb) : "memory");

  for (int k = 0; k < K; k += 32) {
    const int ph = (k >> 5) & 1;
    if (k + 32 < K) {
      const unsigned gk = (unsigned)((k + 32) * 4);
      const unsigned lk = (unsigned)((ph ^ 1) * 2048);
#pragma unroll
      for (int j = 0; j < 4; ++j)
        asm volatile("global_load_async_to_lds_b128 %0, %1, %2"
                     :: "v"(loff[j] + lk), "v"(goff[j] + gk), "s"(bb) : "memory");
      asm volatile("s_wait_asynccnt 0x4" ::: "memory");
    } else {
      asm volatile("s_wait_asynccnt 0x0" ::: "memory");
    }
    // B-fragment from LDS: lane -> column n=lane&15, K half via lane>>4
    const float* p = &stage[wave][ph][(lane & 15) * 32 + (((lane >> 4) & 1) << 3)];
    const v16bf bfr = pack16(*(const v4f*)p, *(const v4f*)(p + 4),
                             *(const v4f*)(p + 16), *(const v4f*)(p + 20));
#pragma unroll
    for (int m = 0; m < 8; ++m)
      acc[m] = wmma_bf16(load_a16x32_bf(outcat_bf + (size_t)m * 16 * K + k, K, lane), bfr, acc[m]);
  }
  const int nl = lane & 15;
  const int mh = (lane >> 4) * 8;
  const int n = nt * 16 + nl;
  const float bias = fc_b[n];
#pragma unroll
  for (int m = 0; m < 8; ++m)
#pragma unroll
    for (int i = 0; i < 8; ++i)
      logits[(size_t)(m * 16 + mh + i) * V + n] = acc[m][i] + bias;
}

extern "C" void kernel_launch(void* const* d_in, const int* in_sizes, int n_in,
                              void* d_out, int out_size, void* d_ws, size_t ws_size,
                              hipStream_t stream) {
  (void)in_sizes; (void)n_in; (void)out_size; (void)ws_size;
  const int*   x      = (const int*)  d_in[0];
  const float* hidden = (const float*)d_in[1];
  const float* cell   = (const float*)d_in[2];
  const float* enc    = (const float*)d_in[3];
  const float* emb    = (const float*)d_in[4];
  const float* attn_W = (const float*)d_in[5];
  const float* attn_b = (const float*)d_in[6];
  const float* vvec   = (const float*)d_in[7];
  const float* W_ih   = (const float*)d_in[8];
  const float* W_hh   = (const float*)d_in[9];
  const float* b_ih   = (const float*)d_in[10];
  const float* b_hh   = (const float*)d_in[11];
  const float* fc_W   = (const float*)d_in[12];
  const float* fc_b   = (const float*)d_in[13];

  float* out      = (float*)d_out;
  float* logits   = out;                          // B*V
  float* h_out    = logits + (size_t)B * V;       // B*H
  float* c_out    = h_out  + (size_t)B * H;       // B*H
  float* attn_out = c_out  + (size_t)B * H;       // B*S

  float* wsf    = (float*)d_ws;
  float* proj   = wsf;                            // B*H
  float* scores = proj   + (size_t)B * H;         // B*S
  float* gates  = scores + (size_t)B * S;         // B*4H
  __bf16* wsb       = (__bf16*)(gates + (size_t)B * 4 * H);
  __bf16* hidden_bf = wsb;                          // B*H
  __bf16* xin_bf    = hidden_bf + (size_t)B * H;    // B*(E+H)
  __bf16* outcat_bf = xin_bf + (size_t)B * (E + H); // B*2H

  dim3 blk(256);
  k_cvt_hidden <<<dim3((B * H + 255) / 256), blk, 0, stream>>>(hidden, hidden_bf);
  k_proj       <<<dim3(8, 8),   blk, 0, stream>>>(hidden, attn_W, attn_b, proj);
  k_score      <<<dim3(B, 8),   blk, 0, stream>>>(enc, attn_W, proj, vvec, scores);
  k_softmax_ctx<<<dim3(B),      blk, 0, stream>>>(scores, enc, x, emb, attn_out, xin_bf, outcat_bf);
  k_gates      <<<dim3(32),     blk, 0, stream>>>(xin_bf, hidden_bf, W_ih, W_hh, b_ih, b_hh, gates);
  k_lstm       <<<dim3((B * H + 255) / 256), blk, 0, stream>>>(gates, cell, h_out, c_out, outcat_bf);
  k_logits     <<<dim3(250),    blk, 0, stream>>>(outcat_bf, fc_W, fc_b, logits);
}
